// BiMamba2DSlow4_18124761989167
// MI455X (gfx1250) — matrix operand order repaired and checked by reading, not compiled, not run on hardware
//
#include <hip/hip_runtime.h>
#include <cstdint>
#include <cstddef>

// ---- problem constants ----
#define DI     192          // d_inner
#define DM     96           // d_model
#define NSTATE 16
#define RK     6            // dt_rank
#define NBATCH 4
#define HDIM   56
#define WDIM   56
#define LSEQ   (HDIM*WDIM)        // 3136
#define MROWS  (NBATCH*LSEQ)      // 12544 rows (b,p)
#define CPROJ  48                 // 38 (dt_rank+2*d_state) padded to 48

typedef __attribute__((ext_vector_type(16))) _Float16     v16h;
typedef __attribute__((ext_vector_type(8)))  float        v8f;
typedef __attribute__((ext_vector_type(4)))  unsigned int v4u;
typedef __attribute__((ext_vector_type(8)))  int          v8i;
typedef __attribute__((ext_vector_type(4)))  int          v4i;

// ---- WMMA operand gathers (CDNA5 ISA 7.12.2, 16-bit A 16x32) ----
// A: lane m=L&15, kh=L>>4; pairs at k = {0,2,4,6,16,18,20,22} + 8*kh
//    -> two contiguous 16B chunks, lowers to 2x ds_load_b128
__device__ __forceinline__ v16h wmma_gather_a(const _Float16* p) {
  v16h x;
#pragma unroll
  for (int v = 0; v < 8; ++v) {
    const int kb = (v < 4) ? (2 * v) : (8 + 2 * v);
    x[2 * v]     = p[kb];
    x[2 * v + 1] = p[kb + 1];
  }
  return x;
}
// B (32x16): lane n=L&15, kgrp=L>>4; K halves {0..7,16..23} + 8*kgrp -> 2x global_load_b128
__device__ __forceinline__ v16h wmma_gather_b(const _Float16* p) {
  v16h x;
#pragma unroll
  for (int i = 0; i < 16; ++i) x[i] = p[(i < 8) ? i : (i + 8)];
  return x;
}

// ---- Tensor Data Mover: stage a 16xK f16 row-major tile into LDS ----
// D# per CDNA5 ISA 8.3-8.5. Treat the tile itself as the tensor (no OOB).
// This toolchain exposes the 6-arg builtin:
//   (uint32x4 g0, int32x8 g1, int32x4 g2, int32x4 g3, int32x8 extra, i32 cpol)
#if __has_builtin(__builtin_amdgcn_tensor_load_to_lds)
#define USE_TDM 1
__device__ __forceinline__ void tdm_load_tile16xK(const _Float16* gptr,
                                                  unsigned int lds_off, int K_) {
  const unsigned long long ga = (unsigned long long)(uintptr_t)gptr;
  v4u g0;
  g0[0] = 1u;                                   // count=1, user descriptor
  g0[1] = lds_off;                              // lds_addr (bytes)
  g0[2] = (unsigned int)ga;                     // global_addr[31:0]
  g0[3] = (unsigned int)((ga >> 32) & 0x01FFFFFFu) | (2u << 30); // addr[56:32]|type=2
  v8i g1;
  g1[0] = 0x00010000;                           // wg_mask=0, data_size=1 (2 bytes)
  g1[1] = (int)(((unsigned int)K_ & 0xFFFFu) << 16);       // tensor_dim0 lo16 @ [63:48]
  g1[2] = (int)((((unsigned int)K_) >> 16) | (16u << 16)); // tdim0 hi16 | tensor_dim1=16
  g1[3] = (int)(((unsigned int)K_ & 0xFFFFu) << 16);       // tdim1 hi=0 | tile_dim0=K
  g1[4] = 16;                                   // tile_dim1=16, tile_dim2=0
  g1[5] = K_;                                   // tensor_dim0_stride[31:0] = K
  g1[6] = 0;                                    // stride hi16 | dim1_stride lo16
  g1[7] = 0;                                    // dim1_stride hi
  v4i z4 = {0, 0, 0, 0};                        // groups 2/3 unused (2D tensor)
  v8i z8 = {0, 0, 0, 0, 0, 0, 0, 0};
  __builtin_amdgcn_tensor_load_to_lds(g0, g1, z4, z4, z8, 0);
}
#else
#define USE_TDM 0
#endif

// ---- generic f16 WMMA GEMM: C[M][N] = A[M][K] * W[N][K]^T, f32 accum ----
// grid.x = M/16 tiles; wave handles ntile = blockIdx.y*(blockDim/32)+wave.
// K is compile-time so the chunk loop fully unrolls: all operand loads are
// hoisted above the WMMA accumulation chain (no per-WMMA load stall).
template <int K>
__global__ void wmma_gemm_kernel(const _Float16* __restrict__ A,
                                 const _Float16* __restrict__ Bw,
                                 float* __restrict__ C, int N) {
  extern __shared__ _Float16 sA[];              // 16 x K tile
  const int mtile = blockIdx.x;
  const int tid   = threadIdx.x;
  const _Float16* gA = A + (size_t)mtile * 16 * K;

#if USE_TDM
  if (tid < 32) {                               // one wave issues the DMA
    tdm_load_tile16xK(gA, (unsigned int)(uintptr_t)(void*)sA, K);
    __builtin_amdgcn_s_wait_tensorcnt(0);
  }
  __syncthreads();
#else
  {
    const uint32_t* gsrc = (const uint32_t*)gA;
    uint32_t* sdst = (uint32_t*)sA;
    for (int i = tid; i < 8 * K; i += blockDim.x) sdst[i] = gsrc[i];
    __syncthreads();
  }
#endif

  const int wave = tid >> 5;
  const int lane = tid & 31;
  const int nt   = blockIdx.y * (blockDim.x >> 5) + wave;
  if (nt * 16 >= N) return;
  const int r  = lane & 15;
  const int kh = lane >> 4;
  const _Float16* browp = Bw + (size_t)(nt * 16 + r) * K + 8 * kh;

  v8f acc = {};
#pragma unroll
  for (int koff = 0; koff < K; koff += 32) {
    v16h a = wmma_gather_a(sA + r * K + koff + 8 * kh);
    v16h b = wmma_gather_b(browp + koff);
    acc = __builtin_amdgcn_wmma_f32_16x16x32_f16(false, a, false, b,
                                                 (short)0, acc, false, false);
  }
#pragma unroll
  for (int v = 0; v < 8; ++v) {
    const int mr = v + 8 * kh;
    C[(size_t)(mtile * 16 + mr) * N + nt * 16 + r] = acc[v];
  }
}

// ---- prep kernels ----
__global__ void cvt_f16_kernel(const float* __restrict__ src,
                               _Float16* __restrict__ dst, int n) {
  int i = blockIdx.x * blockDim.x + threadIdx.x;
  if (i < n) dst[i] = (_Float16)src[i];
}

// pack x_proj_weight (4,38,192) -> (4,48,192) f16, zero-padded rows 38..47
__global__ void prep_wproj_kernel(const float* __restrict__ src,
                                  _Float16* __restrict__ dst) {
  int i = blockIdx.x * blockDim.x + threadIdx.x;   // over 4*48*192
  if (i >= 4 * CPROJ * DI) return;
  int d = i % DI;
  int c = (i / DI) % CPROJ;
  int k = i / (DI * CPROJ);
  float v = (c < 38) ? src[((k * 38) + c) * DI + d] : 0.f;
  dst[i] = (_Float16)v;
}

__global__ void prep_aneg_kernel(const float* __restrict__ Alogs,
                                 float* __restrict__ Aneg, int n) {
  int i = blockIdx.x * blockDim.x + threadIdx.x;
  if (i < n) Aneg[i] = -__expf(Alogs[i]);
}

// ---- depthwise 3x3 conv + bias + SiLU; writes f32 and f16 copies ----
__global__ void conv_silu_kernel(const float* __restrict__ xz,
                                 const float* __restrict__ cw,
                                 const float* __restrict__ cb,
                                 float* __restrict__ xc,
                                 _Float16* __restrict__ xc16) {
  int i = blockIdx.x * blockDim.x + threadIdx.x;
  if (i >= MROWS * DI) return;
  int d  = i % DI;
  int bp = i / DI;
  int b = bp / LSEQ, p = bp % LSEQ;
  int h = p / WDIM,  w = p % WDIM;
  float acc = cb[d];
#pragma unroll
  for (int dy = 0; dy < 3; ++dy) {
    int hh = h + dy - 1;
    if ((unsigned)hh >= (unsigned)HDIM) continue;
#pragma unroll
    for (int dx = 0; dx < 3; ++dx) {
      int ww = w + dx - 1;
      if ((unsigned)ww >= (unsigned)WDIM) continue;
      acc += xz[((size_t)(b * LSEQ + hh * WDIM + ww)) * 384 + d] * cw[d * 9 + dy * 3 + dx];
    }
  }
  float s = acc / (1.f + __expf(-acc));   // SiLU
  xc[i]   = s;
  xc16[i] = (_Float16)s;
}

// ---- selective scan: one block per (batch, direction), one channel/thread ----
__global__ void scan_kernel(const float* __restrict__ proj,  // [MROWS][4*48]
                            const float* __restrict__ xc,    // [MROWS][192]
                            const float* __restrict__ dtw,   // (4,192,6)
                            const float* __restrict__ dtb,   // (4,192)
                            const float* __restrict__ Aneg,  // (4,192,16)
                            const float* __restrict__ Ds,    // (4,192)
                            float* __restrict__ ycomb) {     // [MROWS][192]
  const int b = blockIdx.x;
  const int k = blockIdx.y;
  const int d = threadIdx.x;
  __shared__ float sbc[38];   // [0..5]=dt_raw, [6..21]=B, [22..37]=C

  float wr[RK];
#pragma unroll
  for (int r = 0; r < RK; ++r) wr[r] = dtw[(k * DI + d) * RK + r];
  float Ar[NSTATE];
#pragma unroll
  for (int n = 0; n < NSTATE; ++n) Ar[n] = Aneg[(k * DI + d) * NSTATE + n];
  const float bias = dtb[k * DI + d];
  const float Dd   = Ds[k * DI + d];
  float hst[NSTATE];
#pragma unroll
  for (int n = 0; n < NSTATE; ++n) hst[n] = 0.f;

  for (int l = 0; l < LSEQ; ++l) {
    int q = (k >= 2) ? (LSEQ - 1 - l) : l;
    int p = (k & 1) ? ((q % WDIM) * WDIM + (q / WDIM)) : q;  // transpose for WH scans
    const size_t base = (size_t)(b * LSEQ + p) * DI;
    if (d < 38) sbc[d] = proj[(size_t)(b * LSEQ + p) * (4 * CPROJ) + k * CPROJ + d];
    __syncthreads();

    float u = xc[base + d];
    float draw = bias;
#pragma unroll
    for (int r = 0; r < RK; ++r) draw += sbc[r] * wr[r];
    float dt = (draw > 20.f) ? draw : log1pf(__expf(draw));  // softplus
    float du = dt * u;
    float y = 0.f;
#pragma unroll
    for (int n = 0; n < NSTATE; ++n) {
      hst[n] = __expf(dt * Ar[n]) * hst[n] + du * sbc[6 + n];
      y += hst[n] * sbc[22 + n];
    }
    y += u * Dd;
    atomicAdd(&ycomb[base + d], y);

    // prefetch next step's u row
    if (l + 1 < LSEQ) {
      int q2 = (k >= 2) ? (LSEQ - 2 - l) : (l + 1);
      int p2 = (k & 1) ? ((q2 % WDIM) * WDIM + (q2 / WDIM)) : q2;
      __builtin_prefetch(&xc[(size_t)(b * LSEQ + p2) * DI + d], 0, 0);
    }
    __syncthreads();
  }
}

// ---- LayerNorm over 192 channels + SiLU(z) gate -> f16 for out_proj ----
__global__ void ln_gate_kernel(const float* __restrict__ ycomb,
                               const float* __restrict__ xz,
                               const float* __restrict__ gamma,
                               const float* __restrict__ beta,
                               _Float16* __restrict__ g16) {
  const int bp = blockIdx.x;
  const int d  = threadIdx.x;
  __shared__ float sv[DI];
  __shared__ float s1[64], s2[64];
  float y = ycomb[(size_t)bp * DI + d];
  sv[d] = y;
  __syncthreads();
  if (d < 64) {
    float a0 = sv[d], a1 = sv[d + 64], a2 = sv[d + 128];
    s1[d] = a0 + a1 + a2;
    s2[d] = a0 * a0 + a1 * a1 + a2 * a2;
  }
  __syncthreads();
  for (int off = 32; off > 0; off >>= 1) {
    if (d < off) { s1[d] += s1[d + off]; s2[d] += s2[d + off]; }
    __syncthreads();
  }
  float mean = s1[0] * (1.f / DI);
  float var  = s2[0] * (1.f / DI) - mean * mean;
  float g = (y - mean) * rsqrtf(var + 1e-5f) * gamma[d] + beta[d];
  float z = xz[(size_t)bp * 384 + 192 + d];
  g *= z / (1.f + __expf(-z));
  g16[(size_t)bp * DI + d] = (_Float16)g;
}

extern "C" void kernel_launch(void* const* d_in, const int* in_sizes, int n_in,
                              void* d_out, int out_size, void* d_ws, size_t ws_size,
                              hipStream_t stream) {
  const float* x         = (const float*)d_in[0];   // (4,56,56,96)
  const float* inproj_w  = (const float*)d_in[1];   // (384,96)
  const float* conv_w    = (const float*)d_in[2];   // (192,1,3,3)
  const float* conv_b    = (const float*)d_in[3];   // (192,)
  const float* xproj_w   = (const float*)d_in[4];   // (4,38,192)
  const float* dtw       = (const float*)d_in[5];   // (4,192,6)
  const float* dtb       = (const float*)d_in[6];   // (4,192)
  const float* Alogs     = (const float*)d_in[7];   // (4,192,16)
  const float* Dsp       = (const float*)d_in[8];   // (4,192)
  const float* gamma     = (const float*)d_in[9];   // (192,)
  const float* beta      = (const float*)d_in[10];  // (192,)
  const float* outproj_w = (const float*)d_in[11];  // (96,192)
  float* out = (float*)d_out;

  char* ws = (char*)d_ws;
  size_t off = 0;
  auto alloc = [&](size_t bytes) -> void* {
    void* p = ws + off;
    off = (off + bytes + 255) & ~(size_t)255;
    return p;
  };
  float*    xz     = (float*)    alloc((size_t)MROWS * 384 * 4);  // in_proj out
  float*    xc     = (float*)    alloc((size_t)MROWS * DI * 4);   // conv+silu out
  _Float16* xc16   = (_Float16*) alloc((size_t)MROWS * DI * 2);
  float*    proj   = (float*)    alloc((size_t)MROWS * (4 * CPROJ) * 4);
  float*    ycomb  = (float*)    alloc((size_t)MROWS * DI * 4);
  _Float16* g16    = (_Float16*) alloc((size_t)MROWS * DI * 2);
  _Float16* x16    = (_Float16*) alloc((size_t)MROWS * DM * 2);
  _Float16* w16in  = (_Float16*) alloc((size_t)384 * DM * 2);
  _Float16* w16pr  = (_Float16*) alloc((size_t)4 * CPROJ * DI * 2);
  _Float16* w16out = (_Float16*) alloc((size_t)DM * DI * 2);
  float*    aneg   = (float*)    alloc((size_t)4 * DI * NSTATE * 4);

  // --- weight / input conversion ---
  {
    int n = MROWS * DM;
    cvt_f16_kernel<<<(n + 255) / 256, 256, 0, stream>>>(x, x16, n);
  }
  cvt_f16_kernel<<<(384 * DM + 255) / 256, 256, 0, stream>>>(inproj_w, w16in, 384 * DM);
  cvt_f16_kernel<<<(DM * DI + 255) / 256, 256, 0, stream>>>(outproj_w, w16out, DM * DI);
  prep_wproj_kernel<<<(4 * CPROJ * DI + 255) / 256, 256, 0, stream>>>(xproj_w, w16pr);
  prep_aneg_kernel<<<(4 * DI * NSTATE + 255) / 256, 256, 0, stream>>>(Alogs, aneg, 4 * DI * NSTATE);

  // --- in_proj GEMM: [12544,96] x [384,96]^T -> xz [12544,384] ---
  // 8 waves/block * grid.y=3 -> 24 N-tiles
  wmma_gemm_kernel<DM><<<dim3(MROWS / 16, 3), 256, 16 * DM * 2, stream>>>(x16, w16in, xz, 384);

  // --- depthwise conv + SiLU ---
  {
    int n = MROWS * DI;
    conv_silu_kernel<<<(n + 255) / 256, 256, 0, stream>>>(xz, conv_w, conv_b, xc, xc16);
  }

  // --- fused 4-direction x_proj GEMM: [12544,192] x [192,192]^T -> proj ---
  // 4 waves/block * grid.y=3 -> 12 N-tiles
  wmma_gemm_kernel<DI><<<dim3(MROWS / 16, 3), 128, 16 * DI * 2, stream>>>(xc16, w16pr, proj, 4 * CPROJ);

  // --- selective scans, merged via atomic adds into ycomb ---
  (void)hipMemsetAsync(ycomb, 0, (size_t)MROWS * DI * 4, stream);
  scan_kernel<<<dim3(NBATCH, 4), DI, 0, stream>>>(proj, xc, dtw, dtb, aneg, Dsp, ycomb);

  // --- LayerNorm + gate ---
  ln_gate_kernel<<<MROWS, DI, 0, stream>>>(ycomb, xz, gamma, beta, g16);

  // --- out_proj GEMM: [12544,192] x [96,192]^T -> out [12544,96] ---
  // 6 waves/block * grid.y=1 -> 6 N-tiles
  wmma_gemm_kernel<DI><<<dim3(MROWS / 16, 1), 192, 16 * DI * 2, stream>>>(g16, w16out, out, DM);

  (void)in_sizes; (void)n_in; (void)out_size; (void)ws_size;
}
